// VariationalLinear_28759101014467
// MI455X (gfx1250) — matrix operand order repaired and checked by reading, not compiled, not run on hardware
//
#include <hip/hip_runtime.h>
#include <cstdint>
#include <cstddef>

// Problem geometry (fixed by the reference)
#define IN_DIM   8192
#define OUT_DIM  8192
#define B_ROWS   256
#define KSPLIT   4
#define KSLICE   (IN_DIM / KSPLIT)   // 2048 -> 64 k-steps of 32

typedef __attribute__((ext_vector_type(16))) __bf16 v16bf;
typedef __attribute__((ext_vector_type(4)))  __bf16 v4bf;
typedef __attribute__((ext_vector_type(8)))  float  v8f;

__device__ __forceinline__ float softplus_f(float r) {
  // log1p(exp(r)); guard large r (exp overflow) -> softplus(r) ~= r
  return (r > 15.f) ? r : log1pf(__expf(r));
}

// ---------------------------------------------------------------------------
// Kernel 1: out[b][o] = mu_b[o] + softplus(rho_b[o]) * eps_b[o]
// (seeds the output; GEMM kernel then accumulates on top)
// ---------------------------------------------------------------------------
__global__ void __launch_bounds__(256)
bias_init_kernel(const float* __restrict__ mub,
                 const float* __restrict__ rhob,
                 const float* __restrict__ epsb,
                 float* __restrict__ out, int total) {
  int idx = blockIdx.x * blockDim.x + threadIdx.x;
  if (idx >= total) return;
  int o = idx & (OUT_DIM - 1);
  out[idx] = mub[o] + softplus_f(rhob[o]) * epsb[o];
}

// ---------------------------------------------------------------------------
// Kernel 2: convert x (f32) -> bf16 once into workspace (4 MB).
// ---------------------------------------------------------------------------
__global__ void __launch_bounds__(256)
xcvt_kernel(const float* __restrict__ x, __bf16* __restrict__ xb, int total4) {
  int idx = blockIdx.x * blockDim.x + threadIdx.x;   // one float4 per thread
  if (idx >= total4) return;
  const float4 v = ((const float4*)x)[idx];
  v4bf o;
  o[0] = (__bf16)v.x;
  o[1] = (__bf16)v.y;
  o[2] = (__bf16)v.z;
  o[3] = (__bf16)v.w;
  ((v4bf*)xb)[idx] = o;
}

// ---------------------------------------------------------------------------
// Kernel 3: WMMA GEMM with on-the-fly weight materialization.
// Wave layout: each wave owns one 16-wide OUT column group, ALL 256 batch
// rows (16 M-tiles, 128 accumulator VGPRs), and a K-slice of 2048.
//   waves = (OUT/16) * KSPLIT = 2048  -> grid 256 blocks x 256 threads.
// Per half-k-step: 16 b128 loads fill 8 fragment buffers, an empty asm
// operand-pin forces all 8 to be simultaneously live (defeating the
// scheduler's load->wait0->wmma serialization), then 8 WMMAs issue
// back-to-back behind a single s_wait_loadcnt.
// ---------------------------------------------------------------------------
__global__ void __launch_bounds__(256, 1)
vlinear_wmma_kernel(const __bf16* __restrict__ xb,
                    const float* __restrict__ muw,
                    const float* __restrict__ rhow,
                    const float* __restrict__ epsw,
                    float* __restrict__ out) {
  const unsigned lane = threadIdx.x & 31u;
  const unsigned wave = (blockIdx.x * blockDim.x + threadIdx.x) >> 5;

  const unsigned n0    = (wave & 511u) * 16u;        // OUT column base
  const unsigned kbase = (wave >> 9) * (unsigned)KSLICE;

  // Lane mapping shared by A and B fragments (wave32, 16x16 tiles)
  const unsigned lrow   = lane & 15u;   // row within tile (A: M-row, B: N-col)
  const unsigned hi     = lane >> 4;    // half-wave select
  const unsigned khalfB = hi * 16u;     // B: lanes 16..31 hold K=16..31
  const unsigned koA    = hi * 16u;     // A: lanes 16..31 hold K=16..31

  v8f acc[16];
#pragma unroll
  for (int m = 0; m < 16; ++m)
    acc[m] = v8f{0.f, 0.f, 0.f, 0.f, 0.f, 0.f, 0.f, 0.f};

  // Each lane streams 16 contiguous f32 per matrix per k-step from row n0+lrow
  const size_t wRowBase = (size_t)(n0 + lrow) * IN_DIM + khalfB;
  // A base for this lane (per-tile offsets become instruction immediates)
  const __bf16* aBase = xb + (size_t)lrow * IN_DIM + koA;

  for (unsigned kk = 0; kk < KSLICE; kk += 32u) {
    const unsigned k0 = kbase + kk;

    // ---- Build B fragment: bf16(mu + softplus(rho)*eps), WMMA B layout ----
    const size_t wOff = wRowBase + k0;
    const float4* pm = (const float4*)(muw  + wOff);
    const float4* pr = (const float4*)(rhow + wOff);
    const float4* pe = (const float4*)(epsw + wOff);

    float4 m4[4], r4[4], e4[4];
#pragma unroll
    for (int i = 0; i < 4; ++i) { m4[i] = pm[i]; r4[i] = pr[i]; e4[i] = pe[i]; }

    // Prefetch the weight streams 16 k-steps ahead (speculative; safe at tail)
    __builtin_prefetch(muw  + wOff + 512, 0, 1);
    __builtin_prefetch(rhow + wOff + 512, 0, 1);
    __builtin_prefetch(epsw + wOff + 512, 0, 1);

    v16bf bfrag;
#pragma unroll
    for (int i = 0; i < 4; ++i) {
      bfrag[4 * i + 0] = (__bf16)(m4[i].x + softplus_f(r4[i].x) * e4[i].x);
      bfrag[4 * i + 1] = (__bf16)(m4[i].y + softplus_f(r4[i].y) * e4[i].y);
      bfrag[4 * i + 2] = (__bf16)(m4[i].z + softplus_f(r4[i].z) * e4[i].z);
      bfrag[4 * i + 3] = (__bf16)(m4[i].w + softplus_f(r4[i].w) * e4[i].w);
    }

    // ---- 16 M-tiles in 2 halves of 8: batch loads, pin, then 8 WMMAs -----
    const __bf16* aK = aBase + k0;
#pragma unroll
    for (int half = 0; half < 2; ++half) {
      union AU { v16bf v; uint4 q[2]; } au[8];
#pragma unroll
      for (int t = 0; t < 8; ++t) {
        const size_t aOff = (size_t)((half * 8 + t) * 16) * IN_DIM;
        au[t].q[0] = *(const uint4*)(aK + aOff);        // K = koA .. koA+7
        au[t].q[1] = *(const uint4*)(aK + aOff + 8);    // K = koA+8 .. koA+15
      }
      // Operand pin: forces all 8 fragments simultaneously live and loaded
      // (single s_wait_loadcnt) before any WMMA consumes them.
      asm volatile(""
                   : "+v"(au[0].v), "+v"(au[1].v), "+v"(au[2].v), "+v"(au[3].v),
                     "+v"(au[4].v), "+v"(au[5].v), "+v"(au[6].v), "+v"(au[7].v));
#pragma unroll
      for (int t = 0; t < 8; ++t) {
        acc[half * 8 + t] = __builtin_amdgcn_wmma_f32_16x16x32_bf16(
            /*neg_a=*/false, au[t].v, /*neg_b=*/false, bfrag,
            /*c_mod=*/(short)0, acc[half * 8 + t],
            /*reuse_a=*/false, /*reuse_b=*/false);
      }
    }
  }

  // ---- Accumulate partials into out (bias already seeded) ----
  // C layout: VGPR r, lanes 0-15 -> M = 16*m + r, lanes 16-31 -> M = 16*m + 8 + r
  const unsigned col = n0 + lrow;
#pragma unroll
  for (int m = 0; m < 16; ++m) {
    const unsigned rbase = (unsigned)m * 16u + hi * 8u;
#pragma unroll
    for (int r = 0; r < 8; ++r) {
      unsafeAtomicAdd(out + (size_t)(rbase + r) * OUT_DIM + col, acc[m][r]);
    }
  }
}

// ---------------------------------------------------------------------------
extern "C" void kernel_launch(void* const* d_in, const int* in_sizes, int n_in,
                              void* d_out, int out_size, void* d_ws, size_t ws_size,
                              hipStream_t stream) {
  (void)in_sizes; (void)n_in; (void)ws_size; (void)out_size;

  // setup_inputs() order: x, mu_weights, rho_weights, mu_bias, rho_bias, eps_w, eps_b
  const float* x    = (const float*)d_in[0];
  const float* muw  = (const float*)d_in[1];
  const float* rhow = (const float*)d_in[2];
  const float* mub  = (const float*)d_in[3];
  const float* rhob = (const float*)d_in[4];
  const float* epsw = (const float*)d_in[5];
  const float* epsb = (const float*)d_in[6];
  float* out = (float*)d_out;

  __bf16* xb = (__bf16*)d_ws;                        // 256*8192*2 = 4 MB

  const int total = B_ROWS * OUT_DIM;                // 2M elements
  bias_init_kernel<<<total / 256, 256, 0, stream>>>(mub, rhob, epsb, out, total);

  const int total4 = (B_ROWS * IN_DIM) / 4;          // 512K float4 chunks
  xcvt_kernel<<<total4 / 256, 256, 0, stream>>>(x, xb, total4);

  // 2048 waves: 256 blocks x 256 threads (8 waves/block)
  vlinear_wmma_kernel<<<256, 256, 0, stream>>>(xb, muw, rhow, epsw, out);
}
